// TinyVQVAE_19713899889456
// MI455X (gfx1250) — matrix operand order repaired and checked by reading, compiled, not run on hardware
//
#include <hip/hip_runtime.h>
#include <hip/hip_bf16.h>

typedef __attribute__((ext_vector_type(16))) _Float16 v16h;
typedef __attribute__((ext_vector_type(8)))  _Float16 v8h;
typedef __attribute__((ext_vector_type(8)))  float    v8f;
typedef __attribute__((ext_vector_type(4)))  int      i32x4;

#define Bn 16
#define Hn 128
#define Wn 128
#define Cn 128
#define Kn 512

// ---- CDNA5 async global->LDS path (guarded: falls back to ds stores) ----
#if defined(__HIP_DEVICE_COMPILE__) && \
    __has_builtin(__builtin_amdgcn_global_load_async_to_lds_b128) && \
    __has_builtin(__builtin_amdgcn_s_wait_asynccnt)
#define USE_ASYNC_LDS 1
#else
#define USE_ASYNC_LDS 0
#endif

#if USE_ASYNC_LDS
// builtin takes typed int4* pointers in AS1 (global) / AS3 (LDS).
__device__ __forceinline__ void async_g2l_16B(const void* gsrc, void* ldst) {
  __builtin_amdgcn_global_load_async_to_lds_b128(
      (__attribute__((address_space(1))) i32x4*)(i32x4*)const_cast<void*>(gsrc),
      (__attribute__((address_space(3))) i32x4*)(i32x4*)ldst, 0, 0);
}
#endif

// =====================================================================
// Weight prep: f32 -> f16, conv-transpose flip/swap, codebook norms,
// zero counts/loss accumulators (must happen every call).
// =====================================================================
__global__ void prep_kernel(const float* __restrict__ enc_wh,
                            const float* __restrict__ dec_wh,
                            const float* __restrict__ pre_w,
                            const float* __restrict__ cbf,
                            _Float16* __restrict__ wenc,
                            _Float16* __restrict__ wdec,
                            _Float16* __restrict__ wpre,
                            _Float16* __restrict__ cbh,
                            float* __restrict__ enorm,
                            unsigned int* __restrict__ counts,
                            float* __restrict__ lossSum)
{
  const size_t TOT = 967168;  // 442368 + 442368 + 16384 + 65536 + 512
  for (size_t i = (size_t)blockIdx.x * blockDim.x + threadIdx.x; i < TOT;
       i += (size_t)gridDim.x * blockDim.x) {
    if (i < 442368) {  // encoder hidden convs: wenc[L][tap][o][c] = enc_wh[L][o][c][ky][kx]
      size_t r = i;
      const int L   = (int)(r / 147456); r %= 147456;
      const int tap = (int)(r / 16384);  r %= 16384;
      const int o   = (int)(r / 128);
      const int c   = (int)(r % 128);
      wenc[i] = (_Float16)enc_wh[(((size_t)L * 128 + o) * 128 + c) * 9 + tap];
    } else if (i < 884736) {  // decoder: flip spatial, swap I/O
      const size_t j = i - 442368;
      size_t r = j;
      const int L   = (int)(r / 147456); r %= 147456;
      const int tap = (int)(r / 16384);  r %= 16384;
      const int o   = (int)(r / 128);
      const int c   = (int)(r % 128);
      const int ky = tap / 3, kx = tap % 3;
      const int ftap = (2 - ky) * 3 + (2 - kx);
      wdec[j] = (_Float16)dec_wh[(((size_t)L * 128 + c) * 128 + o) * 9 + ftap];
    } else if (i < 901120) {  // pre-VQ 1x1 [o][c]
      const size_t j = i - 884736;
      wpre[j] = (_Float16)pre_w[j];
    } else if (i < 966656) {  // codebook f16
      const size_t j = i - 901120;
      cbh[j] = (_Float16)cbf[j];
    } else {                  // per-code squared norms (of f16-rounded values)
      const int k = (int)(i - 966656);
      float s = 0.f;
      for (int d = 0; d < 128; ++d) {
        const float v = (float)(_Float16)cbf[(size_t)k * 128 + d];
        s += v * v;
      }
      enorm[k]  = s;
      counts[k] = 0u;
      if (k == 0) lossSum[0] = 0.f;
    }
  }
}

// =====================================================================
// First conv: 3 -> 128 channels, f32 NCHW in, f16 NHWC out. Tiny FLOPs.
// =====================================================================
__global__ __launch_bounds__(256) void conv0_kernel(
    const float* __restrict__ x, const float* __restrict__ w0,
    const float* __restrict__ b0, _Float16* __restrict__ out)
{
  const size_t id = (size_t)blockIdx.x * 256 + threadIdx.x;  // 33,554,432
  const int o = (int)(id & 127);
  const size_t pix = id >> 7;
  const int bb = (int)(pix >> 14);
  const int y  = (int)((pix >> 7) & 127);
  const int xx = (int)(pix & 127);
  float s = b0[o];
  #pragma unroll
  for (int c = 0; c < 3; ++c)
    #pragma unroll
    for (int ky = 0; ky < 3; ++ky) {
      const int gy = y + ky - 1;
      if (gy < 0 || gy >= Hn) continue;
      #pragma unroll
      for (int kx = 0; kx < 3; ++kx) {
        const int gx = xx + kx - 1;
        if (gx < 0 || gx >= Wn) continue;
        s += x[((size_t)(bb * 3 + c) * Hn + gy) * Wn + gx] *
             w0[((o * 3 + c) * 3 + ky) * 3 + kx];
      }
    }
  out[pix * Cn + o] = (_Float16)s;
}

// =====================================================================
// 128->128 conv (KS=3, pad 1) or 1x1 (KS=1) as implicit GEMM via WMMA.
// One workgroup (8 waves) per image row: M=128 pixels, N=128, K=KS*KS*128.
// A fragment per ISA 16-bit A 16x32 layout; B per 32x16 layout from
// pre-transposed weights wp[tap][o][c]. B fragments for a k-step are
// batch-preloaded (one load clause) before the 8-WMMA chain so the
// scheduler can overlap loads with matrix issue.
// =====================================================================
template<int KS>
__global__ __launch_bounds__(256) void conv_wmma_kernel(
    const _Float16* __restrict__ in, const _Float16* __restrict__ wp,
    const float* __restrict__ bias, _Float16* __restrict__ out)
{
  constexpr int LW  = (KS == 3) ? (Wn + 2) : Wn;  // pixels per LDS row (with halo)
  constexpr int PIX = 72;                         // 64 ch + 8-half pad per pixel
  __shared__ _Float16 tile[KS * LW * PIX];        // 56,160 B for KS=3

  const int tid  = threadIdx.x;
  const int lane = tid & 31;
  const int wv   = tid >> 5;
  const int lmod = lane & 15;
  const int hi   = lane >> 4;           // lane group (ISA K-offset select)

  const int blk = blockIdx.x;           // (b*Hn + y)
  const int b = blk / Hn;
  const int y = blk % Hn;
  const size_t rowpix = (size_t)blk * Wn;

  v8f acc[8] = {};

  for (int ch = 0; ch < 2; ++ch) {      // channel halves (64 each) to bound LDS
    __syncthreads();
    // stage KS input rows x LW pixels x 64 channels into LDS (16B chunks)
    for (int i = tid; i < KS * LW * 8; i += 256) {
      const int r  = i / (LW * 8);
      const int pr = i % (LW * 8);
      const int p  = pr >> 3;
      const int cc = pr & 7;
      const int gy = (KS == 3) ? (y + r - 1) : y;
      const int gx = (KS == 3) ? (p - 1)     : p;
      _Float16* dst = &tile[(r * LW + p) * PIX + cc * 8];
      if (gy >= 0 && gy < Hn && gx >= 0 && gx < Wn) {
        const _Float16* src =
            in + (((size_t)(b * Hn + gy) * Wn + gx) * Cn + ch * 64 + cc * 8);
#if USE_ASYNC_LDS
        async_g2l_16B(src, dst);
#else
        *(v8h*)dst = *(const v8h*)src;
#endif
      } else {
        v8h zz = {};
        *(v8h*)dst = zz;
      }
    }
#if USE_ASYNC_LDS
    __builtin_amdgcn_s_wait_asynccnt(0);
#endif
    __syncthreads();

    const int xbase = wv * 16;
    const int klo   = hi * 8;
    for (int tap = 0; tap < KS * KS; ++tap) {
      const int ky = tap / KS;
      const int kx = tap % KS;
      for (int kc = 0; kc < 2; ++kc) {  // two K=32 steps per channel half
        // A: lane row M = xbase+lmod; halves 0-7 -> K=klo.., halves 8-15 -> K=klo+16..
        const _Float16* ap =
            &tile[((ky * LW) + xbase + lmod + ((KS == 3) ? kx : 0)) * PIX +
                  kc * 32 + klo];
        v16h a;
        ((v8h*)&a)[0] = *(const v8h*)ap;
        ((v8h*)&a)[1] = *(const v8h*)(ap + 16);

        // batch-preload all 8 N-block B fragments (single load clause)
        const int kglob = ch * 64 + kc * 32 + hi * 8;
        const _Float16* wrow = wp + (size_t)tap * Cn * Cn + kglob;
        v16h bf[8];
        #pragma unroll
        for (int nb = 0; nb < 8; ++nb) {
          const _Float16* bp = wrow + (size_t)(nb * 16 + lmod) * Cn;
          ((v8h*)&bf[nb])[0] = *(const v8h*)bp;
          ((v8h*)&bf[nb])[1] = *(const v8h*)(bp + 16);
        }
        // 8 back-to-back WMMAs against the preloaded fragments
        #pragma unroll
        for (int nb = 0; nb < 8; ++nb) {
          acc[nb] = __builtin_amdgcn_wmma_f32_16x16x32_f16(
              false, a, false, bf[nb], (short)0, acc[nb], false, false);
        }
      }
    }
  }

  // epilogue: bias + store NHWC f16 (D layout: VGPR v -> M = v + 8*hi)
  const int xm0 = wv * 16 + hi * 8;
  #pragma unroll
  for (int nb = 0; nb < 8; ++nb) {
    const int n = nb * 16 + lmod;
    const float bv = bias[n];
    #pragma unroll
    for (int v = 0; v < 8; ++v) {
      out[(rowpix + xm0 + v) * Cn + n] = (_Float16)(acc[nb][v] + bv);
    }
  }
}

// =====================================================================
// VQ: nearest code via WMMA dot products. score = ||e||^2 - 2 x.e
// (||x||^2 is row-constant -> irrelevant for argmin). Writes quantized
// vectors (straight-through fwd), counts, commitment-loss partials.
// =====================================================================
__global__ __launch_bounds__(256) void vq_kernel(
    const _Float16* __restrict__ z, const _Float16* __restrict__ cb,
    const float* __restrict__ enorm, _Float16* __restrict__ qout,
    unsigned int* __restrict__ counts, float* __restrict__ lossSum)
{
  const int tid  = threadIdx.x;
  const int lane = tid & 31;
  const int wv   = tid >> 5;
  const int lmod = lane & 15;
  const int hi   = lane >> 4;
  const size_t rowpix = (size_t)blockIdx.x * Wn;
  const size_t mypix  = rowpix + wv * 16 + lmod;
  const int klo = hi * 8;

  v16h a[4];
  {
    const _Float16* zp = z + mypix * Cn;
    #pragma unroll
    for (int kc = 0; kc < 4; ++kc) {
      ((v8h*)&a[kc])[0] = *(const v8h*)(zp + kc * 32 + klo);
      ((v8h*)&a[kc])[1] = *(const v8h*)(zp + kc * 32 + klo + 16);
    }
  }

  float best[8];
  int   bestn[8];
  #pragma unroll
  for (int v = 0; v < 8; ++v) { best[v] = 3.4e38f; bestn[v] = 0; }

  for (int ncb = 0; ncb < 32; ++ncb) {   // 32 blocks of 16 codes -> K=512
    const int n = ncb * 16 + lmod;       // this lane's code column
    // batch-preload all 4 K-chunk B fragments, then chain 4 WMMAs
    const _Float16* crow = cb + (size_t)n * 128 + hi * 8;
    v16h bf[4];
    #pragma unroll
    for (int kc = 0; kc < 4; ++kc) {
      const _Float16* bp = crow + kc * 32;
      ((v8h*)&bf[kc])[0] = *(const v8h*)bp;
      ((v8h*)&bf[kc])[1] = *(const v8h*)(bp + 16);
    }
    const float en = enorm[n];
    v8f acc = {};
    #pragma unroll
    for (int kc = 0; kc < 4; ++kc) {
      acc = __builtin_amdgcn_wmma_f32_16x16x32_f16(false, a[kc], false, bf[kc],
                                                   (short)0, acc, false, false);
    }
    #pragma unroll
    for (int v = 0; v < 8; ++v) {
      const float s = en - 2.0f * acc[v];
      if (s < best[v]) { best[v] = s; bestn[v] = n; }
    }
  }

  // butterfly argmin across the 16-lane N group (stays within each half)
  #pragma unroll
  for (int m = 8; m >= 1; m >>= 1) {
    #pragma unroll
    for (int v = 0; v < 8; ++v) {
      const float ob = __shfl_xor(best[v], m, 32);
      const int   on = __shfl_xor(bestn[v], m, 32);
      if (ob < best[v] || (ob == best[v] && on < bestn[v])) {
        best[v] = ob; bestn[v] = on;
      }
    }
  }

  float lsum = 0.f;
  #pragma unroll
  for (int v = 0; v < 8; ++v) {
    const size_t pix = rowpix + wv * 16 + hi * 8 + v;  // VGPR v -> M = v + 8*hi
    const int n = bestn[v];
    const _Float16* crow = cb + (size_t)n * 128 + lmod * 8;
    const _Float16* zr   = z  + pix * Cn + lmod * 8;
    _Float16* qp = qout + pix * Cn + lmod * 8;
    const v8h qv = *(const v8h*)crow;
    const v8h zv = *(const v8h*)zr;
    *(v8h*)qp = qv;
    #pragma unroll
    for (int h = 0; h < 8; ++h) {
      const float d = (float)qv[h] - (float)zv[h];
      lsum += d * d;
    }
    if (lmod == 0) atomicAdd(&counts[n], 1u);
  }
  #pragma unroll
  for (int m = 16; m >= 1; m >>= 1) lsum += __shfl_xor(lsum, m, 32);
  if (lane == 0) atomicAdd(lossSum, lsum);
}

// =====================================================================
// Final conv-transpose 128 -> 3, NHWC f16 in, NCHW f32 out (d_out + 1).
// =====================================================================
__global__ __launch_bounds__(256) void dec_final_kernel(
    const _Float16* __restrict__ hin, const float* __restrict__ wl,
    const float* __restrict__ bl, float* __restrict__ xrec)
{
  const size_t pix = (size_t)blockIdx.x * 256 + threadIdx.x;  // 262144
  const int bb = (int)(pix >> 14);
  const int y  = (int)((pix >> 7) & 127);
  const int xx = (int)(pix & 127);
  float s0 = bl[0], s1 = bl[1], s2 = bl[2];
  for (int ky = 0; ky < 3; ++ky) {
    const int gy = y + ky - 1;
    if (gy < 0 || gy >= Hn) continue;
    for (int kx = 0; kx < 3; ++kx) {
      const int gx = xx + kx - 1;
      if (gx < 0 || gx >= Wn) continue;
      const _Float16* hp = hin + (((size_t)bb * Hn + gy) * Wn + gx) * Cn;
      const int ftap = (2 - ky) * 3 + (2 - kx);
      for (int c = 0; c < 128; ++c) {
        const float hv = (float)hp[c];
        const float* wc = wl + (size_t)c * 27 + ftap;  // wl[c][o][fy][fx]
        s0 += hv * wc[0];
        s1 += hv * wc[9];
        s2 += hv * wc[18];
      }
    }
  }
  xrec[((size_t)(bb * 3 + 0) * Hn + y) * Wn + xx] = s0;
  xrec[((size_t)(bb * 3 + 1) * Hn + y) * Wn + xx] = s1;
  xrec[((size_t)(bb * 3 + 2) * Hn + y) * Wn + xx] = s2;
}

// =====================================================================
// loss + perplexity finalize.
// =====================================================================
__global__ void finalize_kernel(const unsigned int* __restrict__ counts,
                                const float* __restrict__ lossSum,
                                float* __restrict__ out, int out_size)
{
  __shared__ float sh[512];
  const int k = threadIdx.x;
  const float p = (float)counts[k] * (1.0f / 262144.0f);
  sh[k] = p * logf(p + 1e-10f);
  __syncthreads();
  for (int s = 256; s > 0; s >>= 1) {
    if (k < s) sh[k] += sh[k + s];
    __syncthreads();
  }
  if (k == 0) {
    out[out_size - 1] = expf(-sh[0]);
    out[0] = 0.25f * lossSum[0] * (1.0f / (262144.0f * 128.0f));
  }
}

// =====================================================================
extern "C" void kernel_launch(void* const* d_in, const int* in_sizes, int n_in,
                              void* d_out, int out_size, void* d_ws, size_t ws_size,
                              hipStream_t stream)
{
  const float* x      = (const float*)d_in[0];
  const float* enc_w0 = (const float*)d_in[1];
  const float* enc_b0 = (const float*)d_in[2];
  const float* enc_wh = (const float*)d_in[3];
  const float* enc_bh = (const float*)d_in[4];
  const float* pre_w  = (const float*)d_in[5];
  const float* pre_b  = (const float*)d_in[6];
  const float* cbf    = (const float*)d_in[7];
  const float* dec_wh = (const float*)d_in[8];
  const float* dec_bh = (const float*)d_in[9];
  const float* dec_wl = (const float*)d_in[10];
  const float* dec_bl = (const float*)d_in[11];
  float* out = (float*)d_out;

  char* ws = (char*)d_ws;
  const size_t ACT = (size_t)Bn * Hn * Wn * Cn * sizeof(_Float16);  // 64 MB
  _Float16* actA = (_Float16*)(ws);
  _Float16* actB = (_Float16*)(ws + ACT);
  _Float16* wenc = (_Float16*)(ws + 2 * ACT);
  _Float16* wdec = wenc + (size_t)3 * 9 * 128 * 128;
  _Float16* wpre = wdec + (size_t)3 * 9 * 128 * 128;
  _Float16* cbh  = wpre + (size_t)128 * 128;
  float*    enorm   = (float*)(cbh + (size_t)Kn * 128);
  unsigned int* counts = (unsigned int*)(enorm + Kn);
  float*    lossSum = (float*)(counts + Kn);

  prep_kernel<<<3778, 256, 0, stream>>>(enc_wh, dec_wh, pre_w, cbf, wenc, wdec,
                                        wpre, cbh, enorm, counts, lossSum);
  conv0_kernel<<<131072, 256, 0, stream>>>(x, enc_w0, enc_b0, actA);
  conv_wmma_kernel<3><<<2048, 256, 0, stream>>>(actA, wenc + 0 * 147456, enc_bh + 0,   actB);
  conv_wmma_kernel<3><<<2048, 256, 0, stream>>>(actB, wenc + 1 * 147456, enc_bh + 128, actA);
  conv_wmma_kernel<3><<<2048, 256, 0, stream>>>(actA, wenc + 2 * 147456, enc_bh + 256, actB);
  conv_wmma_kernel<1><<<2048, 256, 0, stream>>>(actB, wpre, pre_b, actA);   // z
  vq_kernel<<<2048, 256, 0, stream>>>(actA, cbh, enorm, actB, counts, lossSum);
  conv_wmma_kernel<3><<<2048, 256, 0, stream>>>(actB, wdec + 0 * 147456, dec_bh + 0,   actA);
  conv_wmma_kernel<3><<<2048, 256, 0, stream>>>(actA, wdec + 1 * 147456, dec_bh + 128, actB);
  conv_wmma_kernel<3><<<2048, 256, 0, stream>>>(actB, wdec + 2 * 147456, dec_bh + 256, actA);
  dec_final_kernel<<<1024, 256, 0, stream>>>(actA, dec_wl, dec_bl, out + 1);
  finalize_kernel<<<1, 512, 0, stream>>>(counts, lossSum, out, out_size);
  (void)in_sizes; (void)n_in; (void)ws_size;
}
